// CustomSPNet_35794257445303
// MI455X (gfx1250) — matrix operand order repaired and checked by reading, not compiled, not run on hardware
//
#include <hip/hip_runtime.h>
#include <hip/hip_bf16.h>

typedef __attribute__((ext_vector_type(16))) _Float16 v16h;
typedef __attribute__((ext_vector_type(8)))  _Float16 v8h;
typedef __attribute__((ext_vector_type(8)))  float    v8f;
typedef __attribute__((ext_vector_type(4)))  float    v4f;
typedef __attribute__((ext_vector_type(4)))  int      v4i;

#define HF 512
#define WF 768
#define NPIX (HF*WF)
#define KTOP 2048
#define CAND_MAX 16384

#if __has_builtin(__builtin_amdgcn_global_load_async_to_lds_b128)
#define HAVE_ASYNC_LDS 1
#else
#define HAVE_ASYNC_LDS 0
#endif

__device__ __forceinline__ void wait_async_lds() {
#if __has_builtin(__builtin_amdgcn_s_wait_asynccnt)
  __builtin_amdgcn_s_wait_asynccnt(0);
#else
  asm volatile("s_wait_asynccnt 0" ::: "memory");
#endif
}

// ---------------------------------------------------------------------------
// Fragment loaders (wave32 WMMA 16x16x32 f16; contiguous K order used by this code)
// A 16x32: lanes 0-15 -> M=lane, K 0..15;
//          lanes 16-31 -> same M, K 16..31.
// B 32x16: lanes 0-15 -> N=lane, K 0..15; lanes 16-31 -> same N, K 16..31.
// K (= input channels) is contiguous in both weight pack and NHWC activations,
// so every fragment is exactly two aligned 16-byte loads.
// ---------------------------------------------------------------------------
__device__ __forceinline__ v8h zero8() {
  v8h z;
#pragma unroll
  for (int i = 0; i < 8; ++i) z[i] = (_Float16)0.f;
  return z;
}

__device__ __forceinline__ v16h ld_frag_a(const _Float16* base, bool hi) {
  const _Float16* p = base + (hi ? 16 : 0);
  v8h lo = *(const v8h*)p;
  v8h hh = *(const v8h*)(p + 8);
  return __builtin_shufflevector(lo, hh, 0,1,2,3,4,5,6,7,8,9,10,11,12,13,14,15);
}

__device__ __forceinline__ v16h ld_frag_b(const _Float16* base, bool hi) {
  const _Float16* p = base + (hi ? 16 : 0);
  v8h lo = *(const v8h*)p;
  v8h hh = *(const v8h*)(p + 8);
  return __builtin_shufflevector(lo, hh, 0,1,2,3,4,5,6,7,8,9,10,11,12,13,14,15);
}

// ---------------------------------------------------------------------------
// Weight packing: OIHW f32 -> [tap][OCpad][IC] f16, zero-fill padded OC
// ---------------------------------------------------------------------------
__global__ void k_pack_w(const float* __restrict__ src, _Float16* __restrict__ dst,
                         int OC, int OCpad, int IC, int taps) {
  int i = blockIdx.x * blockDim.x + threadIdx.x;
  int total = taps * OCpad * IC;
  if (i >= total) return;
  int ic = i % IC;
  int rest = i / IC;
  int oc = rest % OCpad;
  int tap = rest / OCpad;
  float v = 0.f;
  if (oc < OC) v = src[(oc * IC + ic) * taps + tap];
  dst[((size_t)tap * OCpad + oc) * IC + ic] = (_Float16)v;
}

// ---------------------------------------------------------------------------
// conv1: 1->64 direct (K=9, not worth WMMA), fused ReLU, NHWC f16 out
// ---------------------------------------------------------------------------
__global__ void k_conv1(const float* __restrict__ img, const float* __restrict__ w,
                        const float* __restrict__ b, _Float16* __restrict__ out) {
  __shared__ float ws[64 * 9];
  __shared__ float bs[64];
  for (int i = threadIdx.x; i < 576; i += blockDim.x) ws[i] = w[i];
  for (int i = threadIdx.x; i < 64; i += blockDim.x) bs[i] = b[i];
  __syncthreads();
  int p = blockIdx.x * blockDim.x + threadIdx.x;
  if (p >= NPIX) return;
  int y = p / WF, x = p % WF;
  float pix[9];
#pragma unroll
  for (int t = 0; t < 9; ++t) {
    int yy = y + t / 3 - 1, xx = x + t % 3 - 1;
    pix[t] = (yy >= 0 && yy < HF && xx >= 0 && xx < WF) ? img[yy * WF + xx] : 0.f;
  }
  _Float16* o = out + (size_t)p * 64;
  for (int oc = 0; oc < 64; ++oc) {
    float acc = bs[oc];
#pragma unroll
    for (int t = 0; t < 9; ++t) acc = fmaf(ws[oc * 9 + t], pix[t], acc);
    o[oc] = (_Float16)fmaxf(acc, 0.f);
  }
}

// ---------------------------------------------------------------------------
// WMMA implicit-GEMM conv (3x3 SAME or 1x1), NHWC f16 in, f16 or f32 out.
// Block = MT x NT waves; each wave owns one 16(OC) x 16(pixel) WMMA tile.
// 3x3 path: input halo tile (3 rows x (16*NT+2) cols x IC) staged to LDS via
// CDNA5 async global->LDS copies on interior tiles (ASYNCcnt +
// s_wait_asynccnt), zero-filled sync copy on boundary tiles. WMMA B-fragments
// are then served from LDS (ds_load_b128 pairs).
// ---------------------------------------------------------------------------
template <int IC, int OCpad, int KH, int MT, int NT, bool RELU, bool OUTF32,
          int H, int W>
__global__ __launch_bounds__(32 * MT * NT)
void k_conv_wmma(const _Float16* __restrict__ in, const _Float16* __restrict__ wp,
                 const float* __restrict__ bias, int realOC,
                 void* __restrict__ outv) {
  constexpr int NTHREADS = 32 * MT * NT;
  constexpr int COLS = 16 * NT + 2;      // pixel tile + 1 halo col each side
  __shared__ _Float16 sB[(KH == 3) ? 3 * COLS * IC : 1];

  const int lane = threadIdx.x & 31;
  const int wid  = threadIdx.x >> 5;
  const int mw = wid % MT, nw = wid / MT;
  const int oc0 = (int)blockIdx.y * (16 * MT) + mw * 16;
  const int nlo = lane & 15;
  const bool hi = lane >= 16;

  v8f acc = {0.f, 0.f, 0.f, 0.f, 0.f, 0.f, 0.f, 0.f};
  const int ocA = oc0 + nlo;
  int pixel;

  if constexpr (KH == 3) {
    constexpr int WB = (W >= 16 * NT) ? (W / (16 * NT)) : 1;  // blocks per row
    const int yrow = (int)blockIdx.x / WB;
    const int xblk = ((int)blockIdx.x % WB) * (16 * NT);
    pixel = yrow * W + xblk + nw * 16 + nlo;

    // ---- stage halo tile rows [yrow-1..yrow+1], cols [xblk-1..xblk+16*NT] ----
    constexpr int CPC = IC / 8;                 // 16B chunks per (row,col)
    constexpr int CHUNKS = 3 * COLS * CPC;
    const bool interior =
        (yrow > 0) && (yrow < H - 1) && (xblk > 0) && (xblk + 16 * NT < W);
    if (interior) {
#if HAVE_ASYNC_LDS
      for (int t = threadIdx.x; t < CHUNKS; t += NTHREADS) {
        int rc = t / CPC, ch = t % CPC;
        int row = rc / COLS, col = rc % COLS;
        const _Float16* g =
            in + ((size_t)((yrow - 1 + row) * W + (xblk - 1 + col))) * IC + ch * 8;
        _Float16* l = &sB[(row * COLS + col) * IC + ch * 8];
        __builtin_amdgcn_global_load_async_to_lds_b128(
            (__attribute__((address_space(1))) v4i*)g,
            (__attribute__((address_space(3))) v4i*)l, 0, 0);
      }
      wait_async_lds();
#else
      for (int t = threadIdx.x; t < CHUNKS; t += NTHREADS) {
        int rc = t / CPC, ch = t % CPC;
        int row = rc / COLS, col = rc % COLS;
        const _Float16* g =
            in + ((size_t)((yrow - 1 + row) * W + (xblk - 1 + col))) * IC + ch * 8;
        *(v8h*)&sB[(row * COLS + col) * IC + ch * 8] = *(const v8h*)g;
      }
#endif
    } else {
      for (int t = threadIdx.x; t < CHUNKS; t += NTHREADS) {
        int rc = t / CPC, ch = t % CPC;
        int row = rc / COLS, col = rc % COLS;
        int yy = yrow - 1 + row, xx = xblk - 1 + col;
        v8h val = zero8();
        if (yy >= 0 && yy < H && xx >= 0 && xx < W)
          val = *(const v8h*)(in + ((size_t)(yy * W + xx)) * IC + ch * 8);
        *(v8h*)&sB[(row * COLS + col) * IC + ch * 8] = val;
      }
    }
    __syncthreads();

    // ---- WMMA main loop: 9 taps x IC/32 chunks, B from LDS ----
#pragma unroll
    for (int tap = 0; tap < 9; ++tap) {
      const int row = tap / 3;
      const int col = nw * 16 + nlo + tap % 3;   // halo-shifted column
      const _Float16* lb = &sB[(row * COLS + col) * IC];
#pragma unroll
      for (int c0 = 0; c0 < IC; c0 += 32) {
        v16h a = ld_frag_a(wp + ((size_t)(tap * OCpad) + ocA) * IC + c0, hi);
        v16h b = ld_frag_b(lb + c0, hi);
        acc = __builtin_amdgcn_wmma_f32_16x16x32_f16(false, a, false, b,
                                                     (short)0, acc, false, false);
      }
    }
  } else {
    const int p0 = (int)blockIdx.x * (16 * NT) + nw * 16;
    pixel = p0 + nlo;
    const _Float16* bb = in + (size_t)pixel * IC;
#pragma unroll
    for (int c0 = 0; c0 < IC; c0 += 32) {
      v16h a = ld_frag_a(wp + (size_t)ocA * IC + c0, hi);
      v16h b = ld_frag_b(bb + c0, hi);
      acc = __builtin_amdgcn_wmma_f32_16x16x32_f16(false, a, false, b,
                                                   (short)0, acc, false, false);
    }
  }

  // Epilogue: C layout -> VGPR v is M=v (+8 for lanes>=16), N=lane&15
  const int ocb = oc0 + (hi ? 8 : 0);
  float vals[8];
#pragma unroll
  for (int v = 0; v < 8; ++v) {
    int oc = ocb + v;
    float bv = (oc < realOC) ? bias[oc] : 0.f;
    float r = acc[v] + bv;
    if (RELU) r = fmaxf(r, 0.f);
    vals[v] = r;
  }
  if (OUTF32) {
    float* o = (float*)outv + (size_t)pixel * OCpad + ocb;
    v4f lo = {vals[0], vals[1], vals[2], vals[3]};
    v4f hh = {vals[4], vals[5], vals[6], vals[7]};
    *(v4f*)o = lo;
    *(v4f*)(o + 4) = hh;
  } else {
    _Float16* o = (_Float16*)outv + (size_t)pixel * OCpad + ocb;
    v8h hv;
#pragma unroll
    for (int v = 0; v < 8; ++v) hv[v] = (_Float16)vals[v];
    *(v8h*)o = hv;
  }
}

// ---------------------------------------------------------------------------
// 2x2 max pool, NHWC f16
// ---------------------------------------------------------------------------
__global__ void k_pool2(const _Float16* __restrict__ in, _Float16* __restrict__ out,
                        int Ho, int Wo, int C) {
  int i = blockIdx.x * blockDim.x + threadIdx.x;
  int total = Ho * Wo * C;
  if (i >= total) return;
  int c = i % C;
  int p = i / C;
  int xo = p % Wo, yo = p / Wo;
  int Wi = Wo * 2;
  const _Float16* b = in + ((size_t)(yo * 2) * Wi + xo * 2) * C + c;
  float m = fmaxf(fmaxf((float)b[0], (float)b[C]),
                  fmaxf((float)b[(size_t)Wi * C], (float)b[(size_t)Wi * C + C]));
  out[i] = (_Float16)m;
}

// ---------------------------------------------------------------------------
// softmax over 65 ch (of 80 padded) + 8x8 pixel shuffle to 512x768 score map
// ---------------------------------------------------------------------------
__global__ void k_softmax_shuffle(const float* __restrict__ lg, float* __restrict__ smap) {
  int cell = blockIdx.x * blockDim.x + threadIdx.x;
  if (cell >= 64 * 96) return;
  const float* l = lg + (size_t)cell * 80;
  float m = l[0];
  for (int i = 1; i < 65; ++i) m = fmaxf(m, l[i]);
  float s = 0.f;
  for (int i = 0; i < 65; ++i) s += __expf(l[i] - m);
  float inv = 1.f / s;
  int hc = cell / 96, wc = cell % 96;
  for (int i = 0; i < 64; ++i) {
    int r = i >> 3, c = i & 7;
    smap[(size_t)(hc * 8 + r) * WF + (wc * 8 + c)] = __expf(l[i] - m) * inv;
  }
}

// ---------------------------------------------------------------------------
// separable 9-wide max pool (SAME) + NMS elementwise stages
// ---------------------------------------------------------------------------
__global__ void k_maxpoolH(const float* __restrict__ in, float* __restrict__ out) {
  int i = blockIdx.x * blockDim.x + threadIdx.x;
  if (i >= NPIX) return;
  int y = i / WF, x = i % WF;
  float m = -3.0e38f;
  for (int d = -4; d <= 4; ++d) {
    int xx = x + d;
    if (xx >= 0 && xx < WF) m = fmaxf(m, in[(size_t)y * WF + xx]);
  }
  out[i] = m;
}
__global__ void k_maxpoolV(const float* __restrict__ in, float* __restrict__ out) {
  int i = blockIdx.x * blockDim.x + threadIdx.x;
  if (i >= NPIX) return;
  int y = i / WF, x = i % WF;
  float m = -3.0e38f;
  for (int d = -4; d <= 4; ++d) {
    int yy = y + d;
    if (yy >= 0 && yy < HF) m = fmaxf(m, in[(size_t)yy * WF + x]);
  }
  out[i] = m;
}
__global__ void k_eqmask(const float* __restrict__ s, const float* __restrict__ mp,
                         float* __restrict__ m) {
  int i = blockIdx.x * blockDim.x + threadIdx.x;
  if (i >= NPIX) return;
  m[i] = (s[i] == mp[i]) ? 1.f : 0.f;
}
__global__ void k_suppressed(const float* __restrict__ s, const float* __restrict__ mpm,
                             float* __restrict__ o) {
  int i = blockIdx.x * blockDim.x + threadIdx.x;
  if (i >= NPIX) return;
  o[i] = (mpm[i] > 0.f) ? 0.f : s[i];
}
__global__ void k_updmask(float* __restrict__ m, const float* __restrict__ mpm,
                          const float* __restrict__ sup, const float* __restrict__ mpsup) {
  int i = blockIdx.x * blockDim.x + threadIdx.x;
  if (i >= NPIX) return;
  bool suppress = mpm[i] > 0.f;
  bool newmax = (sup[i] == mpsup[i]) && !suppress;
  m[i] = ((m[i] != 0.f) || newmax) ? 1.f : 0.f;
}
__global__ void k_final(const float* __restrict__ s, const float* __restrict__ m,
                        float* __restrict__ f) {
  int i = blockIdx.x * blockDim.x + threadIdx.x;
  if (i >= NPIX) return;
  int y = i / WF, x = i % WF;
  float v = (m[i] != 0.f) ? s[i] : 0.f;
  if (y < 4 || y >= HF - 4 || x < 4 || x >= WF - 4) v = -1.f;
  f[i] = v;
}

// ---------------------------------------------------------------------------
// candidate compaction + single-workgroup bitonic sort (128KB LDS) + top-K
// key64 = (float_bits(score) << 32) | (0xFFFFFFFF - idx): sorting key64 desc
// gives score desc, index asc (matches lax.top_k tie-break). Deterministic
// despite atomic compaction order because the sort totally orders (key, idx).
// ---------------------------------------------------------------------------
__global__ void k_init_cand(unsigned long long* __restrict__ cand,
                            unsigned int* __restrict__ cnt) {
  int i = blockIdx.x * blockDim.x + threadIdx.x;
  if (i == 0) *cnt = 0u;
  if (i < CAND_MAX) cand[i] = 0x00000000FFFFFFFFull;  // score 0, idx 0
}
__global__ void k_scan(const float* __restrict__ f, unsigned long long* __restrict__ cand,
                       unsigned int* __restrict__ cnt) {
  int i = blockIdx.x * blockDim.x + threadIdx.x;
  if (i >= NPIX) return;
  float v = f[i];
  if (v > 0.f) {
    unsigned int slot = atomicAdd(cnt, 1u);
    if (slot < CAND_MAX) {
      unsigned int kb = __float_as_uint(v);
      cand[slot] = ((unsigned long long)kb << 32) |
                   (unsigned long long)(0xFFFFFFFFu - (unsigned int)i);
    }
  }
}
__global__ void k_sort_topk(unsigned long long* __restrict__ cand, float* __restrict__ out) {
  extern __shared__ unsigned long long sm[];
  const int N = CAND_MAX;
  const int T = blockDim.x;
  for (int i = threadIdx.x; i < N; i += T) sm[i] = cand[i];
  __syncthreads();
  for (int k = 2; k <= N; k <<= 1) {
    for (int j = k >> 1; j > 0; j >>= 1) {
      for (int i = threadIdx.x; i < N; i += T) {
        int ij = i ^ j;
        if (ij > i) {
          unsigned long long a = sm[i], b = sm[ij];
          bool desc = ((i & k) == 0);
          bool sw = desc ? (a < b) : (a > b);
          if (sw) { sm[i] = b; sm[ij] = a; }
        }
      }
      __syncthreads();
    }
  }
  for (int i = threadIdx.x; i < KTOP; i += T) {
    unsigned long long kv = sm[i];
    unsigned int kb = (unsigned int)(kv >> 32);
    unsigned int idx = 0xFFFFFFFFu - (unsigned int)(kv & 0xFFFFFFFFull);
    out[2 * i + 0] = (float)(idx % (unsigned)WF);
    out[2 * i + 1] = (float)(idx / (unsigned)WF);
    out[2 * KTOP + i] = __uint_as_float(kb);
  }
}

// ---------------------------------------------------------------------------
// descriptor map L2 normalize (per pixel over 256 ch)
// ---------------------------------------------------------------------------
__global__ void k_norm_desc(const float* __restrict__ draw, float* __restrict__ dn) {
  int p = blockIdx.x * blockDim.x + threadIdx.x;
  if (p >= 64 * 96) return;
  const float* a = draw + (size_t)p * 256;
  float* o = dn + (size_t)p * 256;
  float s = 0.f;
  for (int c = 0; c < 256; ++c) s = fmaf(a[c], a[c], s);
  float inv = 1.f / fmaxf(sqrtf(s), 1e-12f);
  for (int c = 0; c < 256; ++c) o[c] = a[c] * inv;
}

// ---------------------------------------------------------------------------
// bilinear descriptor sampling at keypoints + renormalize (block per kpt)
// ---------------------------------------------------------------------------
__global__ void k_sample(const float* __restrict__ outbuf, const float* __restrict__ dn,
                         float* __restrict__ out) {
  int k = blockIdx.x;
  int c = threadIdx.x;
  __shared__ float red[256];
  __shared__ int sx0, sx1, sy0, sy1;
  __shared__ float swx, swy;
  if (c == 0) {
    float kx = outbuf[2 * k], ky = outbuf[2 * k + 1];
    float px = kx - 3.5f, py = ky - 3.5f;
    px /= (96.f * 8.f - 4.5f);
    py /= (64.f * 8.f - 4.5f);
    px = px * 2.f - 1.f;
    py = py * 2.f - 1.f;
    float ix = (px + 1.f) * 0.5f * 95.f;
    float iy = (py + 1.f) * 0.5f * 63.f;
    float x0 = floorf(ix), y0 = floorf(iy);
    swx = ix - x0;
    swy = iy - y0;
    int x0i = min(max((int)x0, 0), 95);
    int y0i = min(max((int)y0, 0), 63);
    sx0 = x0i; sx1 = min(x0i + 1, 95);
    sy0 = y0i; sy1 = min(y0i + 1, 63);
  }
  __syncthreads();
  float wx = swx, wy = swy;
  float d00 = dn[((size_t)sy0 * 96 + sx0) * 256 + c];
  float d01 = dn[((size_t)sy0 * 96 + sx1) * 256 + c];
  float d10 = dn[((size_t)sy1 * 96 + sx0) * 256 + c];
  float d11 = dn[((size_t)sy1 * 96 + sx1) * 256 + c];
  float v = d00 * (1.f - wx) * (1.f - wy) + d01 * wx * (1.f - wy) +
            d10 * (1.f - wx) * wy + d11 * wx * wy;
  red[c] = v * v;
  __syncthreads();
  for (int s2 = 128; s2 > 0; s2 >>= 1) {
    if (c < s2) red[c] += red[c + s2];
    __syncthreads();
  }
  float n = fmaxf(sqrtf(red[0]), 1e-12f);
  out[3 * KTOP + (size_t)k * 256 + c] = v / n;
}

// ---------------------------------------------------------------------------
extern "C" void kernel_launch(void* const* d_in, const int* in_sizes, int n_in,
                              void* d_out, int out_size, void* d_ws, size_t ws_size,
                              hipStream_t stream) {
  const float* image = (const float*)d_in[0];
  const float* w1 = (const float*)d_in[1];  const float* b1 = (const float*)d_in[2];
  const float* w2 = (const float*)d_in[3];  const float* b2 = (const float*)d_in[4];
  const float* w3 = (const float*)d_in[5];  const float* b3 = (const float*)d_in[6];
  const float* w4 = (const float*)d_in[7];  const float* b4 = (const float*)d_in[8];
  const float* w5 = (const float*)d_in[9];  const float* b5 = (const float*)d_in[10];
  const float* w6 = (const float*)d_in[11]; const float* b6 = (const float*)d_in[12];
  const float* w7 = (const float*)d_in[13]; const float* b7 = (const float*)d_in[14];
  const float* w8 = (const float*)d_in[15]; const float* b8 = (const float*)d_in[16];
  const float* ws1 = (const float*)d_in[17]; const float* bs1 = (const float*)d_in[18];
  const float* ws2 = (const float*)d_in[19]; const float* bs2 = (const float*)d_in[20];
  const float* wd1 = (const float*)d_in[21]; const float* bd1 = (const float*)d_in[22];
  const float* wd2 = (const float*)d_in[23]; const float* bd2 = (const float*)d_in[24];
  float* out = (float*)d_out;
  char* ws = (char*)d_ws;

  size_t off = 0;
  auto alloc = [&](size_t bytes) {
    size_t o = off;
    off = (off + bytes + 255) & ~(size_t)255;
    return o;
  };
  _Float16* B0 = (_Float16*)(ws + alloc((size_t)NPIX * 64 * 2));
  _Float16* B1 = (_Float16*)(ws + alloc((size_t)NPIX * 64 * 2));
  float* SML  = (float*)(ws + alloc((size_t)64 * 96 * 80 * 4));
  float* DRAW = (float*)(ws + alloc((size_t)64 * 96 * 256 * 4));
  float* DNRM = (float*)(ws + alloc((size_t)64 * 96 * 256 * 4));
  float* S  = (float*)(ws + alloc((size_t)NPIX * 4));
  float* M  = (float*)(ws + alloc((size_t)NPIX * 4));
  float* T1 = (float*)(ws + alloc((size_t)NPIX * 4));
  float* T2 = (float*)(ws + alloc((size_t)NPIX * 4));
  float* T3 = (float*)(ws + alloc((size_t)NPIX * 4));
  float* T4 = (float*)(ws + alloc((size_t)NPIX * 4));
  unsigned long long* CAND = (unsigned long long*)(ws + alloc((size_t)CAND_MAX * 8));
  unsigned int* CNT = (unsigned int*)(ws + alloc(256));
  _Float16* W2P  = (_Float16*)(ws + alloc((size_t)9 * 64 * 64 * 2));
  _Float16* W3P  = (_Float16*)(ws + alloc((size_t)9 * 64 * 64 * 2));
  _Float16* W4P  = (_Float16*)(ws + alloc((size_t)9 * 64 * 64 * 2));
  _Float16* W5P  = (_Float16*)(ws + alloc((size_t)9 * 128 * 64 * 2));
  _Float16* W6P  = (_Float16*)(ws + alloc((size_t)9 * 128 * 128 * 2));
  _Float16* W7P  = (_Float16*)(ws + alloc((size_t)9 * 128 * 128 * 2));
  _Float16* W8P  = (_Float16*)(ws + alloc((size_t)9 * 128 * 128 * 2));
  _Float16* WS1P = (_Float16*)(ws + alloc((size_t)9 * 256 * 128 * 2));
  _Float16* WS2P = (_Float16*)(ws + alloc((size_t)1 * 80 * 256 * 2));
  _Float16* WD1P = (_Float16*)(ws + alloc((size_t)9 * 256 * 128 * 2));
  _Float16* WD2P = (_Float16*)(ws + alloc((size_t)1 * 256 * 256 * 2));

  auto cdiv = [](int a, int b) { return (a + b - 1) / b; };
  auto pack = [&](const float* src, _Float16* dst, int OC, int OCpad, int IC, int taps) {
    int total = taps * OCpad * IC;
    k_pack_w<<<cdiv(total, 256), 256, 0, stream>>>(src, dst, OC, OCpad, IC, taps);
  };
  pack(w2, W2P, 64, 64, 64, 9);
  pack(w3, W3P, 64, 64, 64, 9);
  pack(w4, W4P, 64, 64, 64, 9);
  pack(w5, W5P, 128, 128, 64, 9);
  pack(w6, W6P, 128, 128, 128, 9);
  pack(w7, W7P, 128, 128, 128, 9);
  pack(w8, W8P, 128, 128, 128, 9);
  pack(ws1, WS1P, 256, 256, 128, 9);
  pack(ws2, WS2P, 65, 80, 256, 1);
  pack(wd1, WD1P, 256, 256, 128, 9);
  pack(wd2, WD2P, 256, 256, 256, 1);

  // ---- backbone ----
  k_conv1<<<cdiv(NPIX, 256), 256, 0, stream>>>(image, w1, b1, B0);

  k_conv_wmma<64, 64, 3, 4, 2, true, false, 512, 768>
      <<<dim3(NPIX / 32, 1), 256, 0, stream>>>(B0, W2P, b2, 64, B1);
  k_pool2<<<cdiv(256 * 384 * 64, 256), 256, 0, stream>>>(B1, B0, 256, 384, 64);

  k_conv_wmma<64, 64, 3, 4, 2, true, false, 256, 384>
      <<<dim3(256 * 384 / 32, 1), 256, 0, stream>>>(B0, W3P, b3, 64, B1);
  k_conv_wmma<64, 64, 3, 4, 2, true, false, 256, 384>
      <<<dim3(256 * 384 / 32, 1), 256, 0, stream>>>(B1, W4P, b4, 64, B0);
  k_pool2<<<cdiv(128 * 192 * 64, 256), 256, 0, stream>>>(B0, B1, 128, 192, 64);

  k_conv_wmma<64, 128, 3, 4, 2, true, false, 128, 192>
      <<<dim3(128 * 192 / 32, 2), 256, 0, stream>>>(B1, W5P, b5, 128, B0);
  k_conv_wmma<128, 128, 3, 4, 2, true, false, 128, 192>
      <<<dim3(128 * 192 / 32, 2), 256, 0, stream>>>(B0, W6P, b6, 128, B1);
  k_pool2<<<cdiv(64 * 96 * 128, 256), 256, 0, stream>>>(B1, B0, 64, 96, 128);

  k_conv_wmma<128, 128, 3, 4, 2, true, false, 64, 96>
      <<<dim3(64 * 96 / 32, 2), 256, 0, stream>>>(B0, W7P, b7, 128, B1);
  k_conv_wmma<128, 128, 3, 4, 2, true, false, 64, 96>
      <<<dim3(64 * 96 / 32, 2), 256, 0, stream>>>(B1, W8P, b8, 128, B0);
  // feat = B0 (64x96x128 f16, NHWC)

  // ---- score head ----
  k_conv_wmma<128, 256, 3, 4, 2, true, false, 64, 96>
      <<<dim3(64 * 96 / 32, 4), 256, 0, stream>>>(B0, WS1P, bs1, 256, B1);
  k_conv_wmma<256, 80, 1, 1, 8, false, true, 64, 96>
      <<<dim3(64 * 96 / 128, 5), 256, 0, stream>>>(B1, WS2P, bs2, 65, SML);
  k_softmax_shuffle<<<cdiv(64 * 96, 256), 256, 0, stream>>>(SML, S);

  // ---- NMS (r=4), 2 refinement iterations ----
  int ewg = cdiv(NPIX, 256);
  k_maxpoolH<<<ewg, 256, 0, stream>>>(S, T2);
  k_maxpoolV<<<ewg, 256, 0, stream>>>(T2, T1);
  k_eqmask<<<ewg, 256, 0, stream>>>(S, T1, M);
  for (int it = 0; it < 2; ++it) {
    k_maxpoolH<<<ewg, 256, 0, stream>>>(M, T2);
    k_maxpoolV<<<ewg, 256, 0, stream>>>(T2, T1);          // T1 = maxpool(mask)
    k_suppressed<<<ewg, 256, 0, stream>>>(S, T1, T2);     // T2 = suppressed scores
    k_maxpoolH<<<ewg, 256, 0, stream>>>(T2, T4);
    k_maxpoolV<<<ewg, 256, 0, stream>>>(T4, T3);          // T3 = maxpool(suppressed)
    k_updmask<<<ewg, 256, 0, stream>>>(M, T1, T2, T3);
  }
  k_final<<<ewg, 256, 0, stream>>>(S, M, T4);             // T4 = final map w/ border

  // ---- top-K ----
  k_init_cand<<<cdiv(CAND_MAX, 256), 256, 0, stream>>>(CAND, CNT);
  k_scan<<<ewg, 256, 0, stream>>>(T4, CAND, CNT);
  k_sort_topk<<<1, 1024, CAND_MAX * 8, stream>>>(CAND, out);

  // ---- descriptor head ----
  k_conv_wmma<128, 256, 3, 4, 2, true, false, 64, 96>
      <<<dim3(64 * 96 / 32, 4), 256, 0, stream>>>(B0, WD1P, bd1, 256, B1);
  k_conv_wmma<256, 256, 1, 4, 2, false, true, 64, 96>
      <<<dim3(64 * 96 / 32, 4), 256, 0, stream>>>(B1, WD2P, bd2, 256, DRAW);
  k_norm_desc<<<cdiv(64 * 96, 256), 256, 0, stream>>>(DRAW, DNRM);
  k_sample<<<KTOP, 256, 0, stream>>>(out, DNRM, out);

  (void)in_sizes; (void)n_in; (void)out_size; (void)ws_size;
}
